// SelfAttentionND_40999757808140
// MI455X (gfx1250) — compile-verified, matrix operand order checked by reading, not run on hardware
//
#include <hip/hip_runtime.h>

typedef _Float16 half_t;
typedef __attribute__((ext_vector_type(16))) _Float16 v16h;
typedef __attribute__((ext_vector_type(8)))  _Float16 v8h;
typedef __attribute__((ext_vector_type(8)))  float    v8f;

#define NWTOK 256
#define CCH   256
#define NHEAD 8
#define SCALE_QK 0.17677669529663689f   // 1/sqrt(32)

// ---- LDS layout (total = 320000 bytes <= 320 KB/WGP) ----
#define XN_OFF   0            // half [256][264]  normalized input (A rows: [token][ch])
#define OALL_OFF 135168       // half [256][264]  attn output (B rows: [token][inner]); reused as LN raw staging
#define QS_OFF   270336       // half [256][32]   Q rows [query][dh]
#define KS_OFF   286720       // half [256][32]   K rows [key][dh]
#define VT_OFF   303104       // half [32][264]   V^T rows [dh][token]
#define LDS_BYTES 320000
#define XSTR 264
#define QSTR 32

// One operand loader for both A and B (B assumed to mirror the documented A
// interleave: lane-half selects K sub-band {0..7,16..23} vs {8..15,24..31}).
// `tile` = first row (A: M row / B: N column) of the 16-wide tile; storage is
// row-major [row][k] with the given stride (halves). Emits 2x b128 loads.
static __device__ __forceinline__ v16h load_ab(const half_t* __restrict__ base,
                                               int stride, int tile, int kbase, int lane)
{
  const int m  = lane & 15;
  const int hi = (lane >> 4) & 1;
  const half_t* p = base + (size_t)(tile + m) * stride + kbase + hi * 8;
  v8h lo = *(const v8h*)(p);
  v8h hh = *(const v8h*)(p + 16);
  v16h r;
#pragma unroll
  for (int i = 0; i < 8; ++i) { r[i] = lo[i]; r[8 + i] = hh[i]; }
  return r;
}

static __device__ __forceinline__ v8f wmma16(v16h a, v16h b, v8f c)
{
  return __builtin_amdgcn_wmma_f32_16x16x32_f16(false, a, false, b, (short)0, c, false, false);
}

// ---------------------------------------------------------------------------
// Prep: f16-transpose weights into B-operand-friendly layouts, expand the
// relative-position bias table to [head][query][key] f16.
// ---------------------------------------------------------------------------
__global__ void __launch_bounds__(256) swin_prep_kernel(
    const float* __restrict__ w_qkv,      // [256][768]
    const float* __restrict__ w_out,      // [256][256]
    const float* __restrict__ bias_table, // [1575][8]
    const int*   __restrict__ rel_index,  // [256][256] (query, key)
    half_t* __restrict__ wqkvT,           // [768][256]  wqkvT[n][k] = w_qkv[k][n]
    half_t* __restrict__ woutT,           // [256][256]  woutT[c][k] = w_out[k][c]
    half_t* __restrict__ biasH)           // [8][256][256]
{
  const int idx = blockIdx.x * 256 + threadIdx.x;
  if (idx < 768 * 256) {
    const int n = idx >> 8, k = idx & 255;
    wqkvT[idx] = (half_t)w_qkv[k * 768 + n];
  } else if (idx < 768 * 256 + 256 * 256) {
    const int i = idx - 768 * 256;
    const int c = i >> 8, k = i & 255;
    woutT[i] = (half_t)w_out[k * 256 + c];
  } else {
    const int i  = idx - (768 * 256 + 256 * 256);   // < 8*65536
    const int hh = i >> 16;
    const int nm = i & 65535;                       // query*256 + key
    biasH[i] = (half_t)bias_table[rel_index[nm] * 8 + hh];
  }
}

// ---------------------------------------------------------------------------
// Fully fused window-attention block: one workgroup (8 wave32) per window.
// ---------------------------------------------------------------------------
__global__ void __launch_bounds__(256) swin_window_attn_kernel(
    const float* __restrict__ x,        // (1,256,16,64,64)
    const float* __restrict__ gamma,
    const float* __restrict__ beta,
    const half_t* __restrict__ wqkvT,   // [768][256]
    const half_t* __restrict__ woutT,   // [256][256]
    const float* __restrict__ b_out,
    const half_t* __restrict__ biasH,   // [8][256][256]
    float* __restrict__ out)            // (1,256,16,64,64)
{
  extern __shared__ char smem[];
  half_t* XN   = (half_t*)(smem + XN_OFF);
  half_t* OALL = (half_t*)(smem + OALL_OFF);
  half_t* QS   = (half_t*)(smem + QS_OFF);
  half_t* KS   = (half_t*)(smem + KS_OFF);
  half_t* VT   = (half_t*)(smem + VT_OFF);
  float*  sGamma = (float*)(smem + QS_OFF);   // overlay while QS unused (LN phase)
  float*  sBeta  = sGamma + 256;
  float*  sBout  = (float*)(smem + QS_OFF);   // overlay while QS unused (out phase)

  const int tid  = threadIdx.x;
  const int lane = tid & 31;
  const int wave = tid >> 5;
  const int hi   = (lane >> 4) & 1;
  const int nsel = lane & 15;

  const int win = blockIdx.x;
  const int gd = win >> 6, gh = (win >> 3) & 7, gw = win & 7;
  const int d0 = gd * 4, h0 = gh * 8, w0 = gw * 8;

  const v8f vzero = {0.f, 0.f, 0.f, 0.f, 0.f, 0.f, 0.f, 0.f};

  // Stage LN affine params into LDS.
  sGamma[tid] = gamma[tid];
  sBeta[tid]  = beta[tid];

  // ---- Phase 1: window gather + LayerNorm (one thread per token) ----
  {
    const int wd = tid >> 6, wh = (tid >> 3) & 7, ww = tid & 7;
    const size_t sp = (size_t)(d0 + wd) * 4096 + (size_t)(h0 + wh) * 64 + (size_t)(w0 + ww);
    const float* xp = x + sp;
    half_t* raw = OALL + (size_t)tid * XSTR;    // OALL region is free right now
    float s = 0.f, ss = 0.f;
#pragma unroll 8
    for (int c = 0; c < CCH; ++c) {
      float v = xp[(size_t)c * 65536];
      s += v; ss += v * v;
      raw[c] = (half_t)v;
    }
    const float mu  = s * (1.f / 256.f);
    const float var = ss * (1.f / 256.f) - mu * mu;
    const float rs  = rsqrtf(var + 1e-5f);
    __syncthreads();                            // gamma/beta staged by all threads
    half_t* xr = XN + (size_t)tid * XSTR;
#pragma unroll 8
    for (int c = 0; c < CCH; ++c) {
      float v = ((float)raw[c] - mu) * rs * sGamma[c] + sBeta[c];
      xr[c] = (half_t)v;
    }
  }
  __syncthreads();

  // ---- Phase 2: per head, QKV projection then attention ----
#pragma unroll 1
  for (int h = 0; h < NHEAD; ++h) {
    // --- QKV GEMM: wave owns token tiles {2w, 2w+1}; 6 output tiles (q,k,v x 2) ---
#pragma unroll 1
    for (int mt = 0; mt < 2; ++mt) {
      const int mrow = (wave * 2 + mt) * 16;
      v8f acc[6];
#pragma unroll
      for (int j = 0; j < 6; ++j) acc[j] = vzero;
      const int nb0 = h * 32;
#pragma unroll
      for (int ks = 0; ks < 8; ++ks) {
        v16h a = load_ab(XN, XSTR, mrow, ks * 32, lane);
#pragma unroll
        for (int j = 0; j < 6; ++j) {
          const int nrow = (j >> 1) * 256 + nb0 + (j & 1) * 16;  // q:0.., k:256.., v:512..
          v16h b = load_ab(wqkvT, 256, nrow, ks * 32, lane);
          acc[j] = wmma16(a, b, acc[j]);
        }
      }
      // scatter D tiles: (m=token r+8*hi, n=dh lane&15)
#pragma unroll
      for (int j = 0; j < 6; ++j) {
        const int dh = (j & 1) * 16 + nsel;
#pragma unroll
        for (int r = 0; r < 8; ++r) {
          const int tok = mrow + r + 8 * hi;
          const half_t v = (half_t)acc[j][r];
          if (j < 2)      QS[tok * QSTR + dh] = v;
          else if (j < 4) KS[tok * QSTR + dh] = v;
          else            VT[dh * XSTR + tok] = v;      // transposed store
        }
      }
    }
    __syncthreads();

    // --- attention: S^T = K·Q^T then O^T = V^T·P^T ; wave owns query tiles {w, w+8} ---
    const half_t* biasbase = biasH + (size_t)h * 65536;
#pragma unroll 1
    for (int p = 0; p < 2; ++p) {
      const int qbase = (wave + 8 * p) * 16;
      const v16h bq = load_ab(QS, QSTR, qbase, 0, lane);     // B: n=query, k=dh(32)
      const half_t* biasp = biasbase + (size_t)(qbase + nsel) * 256 + hi * 8;
      float sreg[16][8];                                     // S^T tiles, this lane: one query
      float vmax = -3.0e38f;
#pragma unroll
      for (int kt = 0; kt < 16; ++kt) {
        v16h ak = load_ab(KS, QSTR, kt * 16, 0, lane);       // A: m=key, k=dh
        v8f sc = wmma16(ak, bq, vzero);
        v8h bb = *(const v8h*)(biasp + kt * 16);             // bias[h][query][key..key+7]
#pragma unroll
        for (int r = 0; r < 8; ++r) {
          float f = sc[r] * SCALE_QK + (float)bb[r];
          sreg[kt][r] = f;
          vmax = fmaxf(vmax, f);
        }
      }
      vmax = fmaxf(vmax, __shfl_xor(vmax, 16));              // lanes l, l+16 share a query
      float rsum = 0.f;
      v8f o0 = vzero, o1 = vzero;
#pragma unroll
      for (int kb = 0; kb < 8; ++kb) {                       // 32-key blocks
        v16h pb;                                             // D tiles ARE the B operand
#pragma unroll
        for (int r = 0; r < 8; ++r) {
          float e0 = __expf(sreg[2 * kb][r] - vmax);
          float e1 = __expf(sreg[2 * kb + 1][r] - vmax);
          rsum += e0 + e1;
          pb[r]     = (half_t)e0;
          pb[8 + r] = (half_t)e1;
        }
        v16h av0 = load_ab(VT, XSTR, 0,  kb * 32, lane);     // A: m=dh 0..15,  k=key
        v16h av1 = load_ab(VT, XSTR, 16, kb * 32, lane);     // A: m=dh 16..31, k=key
        o0 = wmma16(av0, pb, o0);
        o1 = wmma16(av1, pb, o1);
      }
      rsum += __shfl_xor(rsum, 16);
      const float inv = 1.f / rsum;
      half_t* orow = OALL + (size_t)(qbase + nsel) * XSTR + h * 32;
#pragma unroll
      for (int r = 0; r < 8; ++r) {                          // O^T: n=query=lane&15 too
        orow[r + 8 * hi]      = (half_t)(o0[r] * inv);
        orow[16 + r + 8 * hi] = (half_t)(o1[r] * inv);
      }
    }
    __syncthreads();
  }

  // ---- Phase 3: output projection out^T = woutT · OALL^T (+b_out) + window merge ----
  sBout[tid] = b_out[tid];
  __syncthreads();
#pragma unroll 1
  for (int mt = 0; mt < 2; ++mt) {
    const int crow = (wave * 2 + mt) * 16;                   // channel tile
#pragma unroll 1
    for (int ch = 0; ch < 2; ++ch) {
      v8f acc[8];
#pragma unroll
      for (int j = 0; j < 8; ++j) acc[j] = vzero;
#pragma unroll
      for (int ks = 0; ks < 8; ++ks) {
        v16h a = load_ab(woutT, 256, crow, ks * 32, lane);   // A: m=channel, k=inner
#pragma unroll
        for (int j = 0; j < 8; ++j) {
          v16h b = load_ab(OALL, XSTR, (ch * 8 + j) * 16, ks * 32, lane); // B: n=token
          acc[j] = wmma16(a, b, acc[j]);
        }
      }
#pragma unroll
      for (int j = 0; j < 8; ++j) {
        const int tok = (ch * 8 + j) * 16 + nsel;
        const int wd = tok >> 6, wh = (tok >> 3) & 7, ww = tok & 7;
        const size_t sp = (size_t)(d0 + wd) * 4096 + (size_t)(h0 + wh) * 64 + (size_t)(w0 + ww);
#pragma unroll
        for (int r = 0; r < 8; ++r) {
          const int c = crow + r + 8 * hi;
          out[(size_t)c * 65536 + sp] = acc[j][r] + sBout[c];  // lanes 0..7: 32B runs
        }
      }
    }
  }
}

// ---------------------------------------------------------------------------
extern "C" void kernel_launch(void* const* d_in, const int* in_sizes, int n_in,
                              void* d_out, int out_size, void* d_ws, size_t ws_size,
                              hipStream_t stream)
{
  (void)in_sizes; (void)n_in; (void)out_size; (void)ws_size;
  const float* x      = (const float*)d_in[0];
  const float* gamma  = (const float*)d_in[1];
  const float* beta   = (const float*)d_in[2];
  const float* w_qkv  = (const float*)d_in[3];
  const float* w_out  = (const float*)d_in[4];
  const float* b_out  = (const float*)d_in[5];
  const float* btable = (const float*)d_in[6];
  const int*   relidx = (const int*)d_in[7];
  float* out = (float*)d_out;

  // Workspace: only 1.57 MB of f16 weight/bias staging.
  half_t* wqkvT = (half_t*)d_ws;            // 768*256
  half_t* woutT = wqkvT + 768 * 256;        // 256*256
  half_t* biasH = woutT + 256 * 256;        // 8*256*256

  swin_prep_kernel<<<3072, 256, 0, stream>>>(w_qkv, w_out, btable, relidx,
                                             wqkvT, woutT, biasH);

  (void)hipFuncSetAttribute((const void*)swin_window_attn_kernel,
                            hipFuncAttributeMaxDynamicSharedMemorySize, LDS_BYTES);
  swin_window_attn_kernel<<<256, 256, LDS_BYTES, stream>>>(
      x, gamma, beta, wqkvT, woutT, b_out, biasH, out);
}
